// CustomMoE_80891414053578
// MI455X (gfx1250) — matrix operand order repaired and checked by reading, not compiled, not run on hardware
//
#include <hip/hip_runtime.h>

typedef __attribute__((ext_vector_type(16))) __bf16 v16bf;
typedef __attribute__((ext_vector_type(8)))  __bf16 v8bf;
typedef __attribute__((ext_vector_type(8)))  float  v8f;

#define N_TOK 8192
#define MD    2048
#define HID   4096
#define NE    8
#define CAP   2048   // k*N/E

// ---- CDNA5 async global->LDS path (guarded; falls back to sync copies) ----
#if defined(__has_builtin)
#if __has_builtin(__builtin_amdgcn_global_load_async_to_lds_b128) && \
    __has_builtin(__builtin_amdgcn_s_wait_asynccnt)
#define MOE_ASYNC 1
#endif
#endif
#ifndef MOE_ASYNC
#define MOE_ASYNC 0
#endif

#define AS1 __attribute__((address_space(1)))
#define AS3 __attribute__((address_space(3)))

// The builtin's params are gcc-vector int4 pointers in AS1 (global) / AS3 (LDS).
typedef int moe_v4i __attribute__((vector_size(16)));

__device__ __forceinline__ AS1 moe_v4i* gcast(const void* p) {
    return (AS1 moe_v4i*)(unsigned long long)p;          // global VA == generic VA
}
__device__ __forceinline__ AS3 moe_v4i* lcast(const void* p) {
    // generic LDS address = aperture | offset; low 32 bits are the LDS offset
    return (AS3 moe_v4i*)(unsigned int)(unsigned long long)p;
}

// ---------------------------------------------------------------- gate ------
// One wave (32 lanes) per token: scores = softmax(x @ wg), top-2, gates.
__global__ __launch_bounds__(256) void moe_gate(const float* __restrict__ x,
                                                const float* __restrict__ wg,
                                                float* __restrict__ scores,
                                                int* __restrict__ tki,
                                                float* __restrict__ gates) {
    const int wave = threadIdx.x >> 5, lane = threadIdx.x & 31;
    const int n = blockIdx.x * 8 + wave;
    const float* xr = x + (size_t)n * MD;
    float acc[NE];
#pragma unroll
    for (int e = 0; e < NE; ++e) acc[e] = 0.f;
    for (int d = lane; d < MD; d += 32) {
        float xv = xr[d];
        const float4* w4 = (const float4*)(wg + (size_t)d * NE);
        float4 a = w4[0], b = w4[1];
        acc[0] += xv * a.x; acc[1] += xv * a.y; acc[2] += xv * a.z; acc[3] += xv * a.w;
        acc[4] += xv * b.x; acc[5] += xv * b.y; acc[6] += xv * b.z; acc[7] += xv * b.w;
    }
#pragma unroll
    for (int off = 16; off > 0; off >>= 1)
#pragma unroll
        for (int e = 0; e < NE; ++e) acc[e] += __shfl_xor(acc[e], off, 32);
    if (lane == 0) {
        float mx = acc[0];
#pragma unroll
        for (int e = 1; e < NE; ++e) mx = fmaxf(mx, acc[e]);
        float p[NE], s = 0.f;
#pragma unroll
        for (int e = 0; e < NE; ++e) { p[e] = __expf(acc[e] - mx); s += p[e]; }
        float inv = 1.f / s;
#pragma unroll
        for (int e = 0; e < NE; ++e) { p[e] *= inv; scores[(size_t)n * NE + e] = p[e]; }
        int i0 = 0;
#pragma unroll
        for (int e = 1; e < NE; ++e) if (p[e] > p[i0]) i0 = e;
        int i1 = (i0 == 0) ? 1 : 0;
#pragma unroll
        for (int e = 0; e < NE; ++e) if (e != i0 && p[e] > p[i1]) i1 = e;
        float g0 = p[i0], g1 = p[i1], gs = g0 + g1;
        tki[n * 2] = i0; tki[n * 2 + 1] = i1;
        gates[n * 2] = g0 / gs; gates[n * 2 + 1] = g1 / gs;
    }
}

// ------------------------------------------------- aux-loss stats (determ) --
__global__ __launch_bounds__(256) void moe_stats(const float* __restrict__ scores,
                                                 const int* __restrict__ tki,
                                                 float* __restrict__ me,
                                                 float* __restrict__ ce) {
    const int e = blockIdx.x, t = threadIdx.x;
    float sm = 0.f, sc = 0.f;
    for (int n = t; n < N_TOK; n += 256) {
        sm += scores[(size_t)n * NE + e];
        sc += (tki[n * 2] == e) ? 1.f : 0.f;
    }
    __shared__ float ssm[256], ssc[256];
    ssm[t] = sm; ssc[t] = sc; __syncthreads();
    for (int off = 128; off > 0; off >>= 1) {
        if (t < off) { ssm[t] += ssm[t + off]; ssc[t] += ssc[t + off]; }
        __syncthreads();
    }
    if (t == 0) { me[e] = ssm[0] / N_TOK; ce[e] = ssc[0] / N_TOK; }
}

__global__ void moe_laux(const float* __restrict__ me, const float* __restrict__ ce,
                         float* __restrict__ out_scalar) {
    if (threadIdx.x == 0) {
        float s = 0.f;
#pragma unroll
        for (int e = 0; e < NE; ++e) s += me[e] * ce[e];
        *out_scalar = (float)NE * s;
    }
}

// --------------------------------------- slot-major capacity positions ------
__global__ __launch_bounds__(256) void moe_pos(const int* __restrict__ tki,
                                               int* __restrict__ flatq) {
    const int e = blockIdx.x, t = threadIdx.x;
    __shared__ int sc[256];
    int run = 0;
    for (int chunk = 0; chunk < (2 * N_TOK) / 256; ++chunk) {
        int idx = chunk * 256 + t;
        int s = idx >> 13;            // slot (0..1), slot-major ordering
        int n = idx & (N_TOK - 1);
        int flag = (tki[n * 2 + s] == e) ? 1 : 0;
        sc[t] = flag; __syncthreads();
        for (int off = 1; off < 256; off <<= 1) {
            int v = (t >= off) ? sc[t - off] : 0;
            __syncthreads();
            sc[t] += v;
            __syncthreads();
        }
        int incl = sc[t];
        int tot = sc[255];
        if (flag) {
            int pos = run + incl - 1;
            flatq[n * 2 + s] = (pos < CAP) ? (e * CAP + pos) : -1;
        }
        run += tot;
        __syncthreads();
    }
}

// ------------------------------------------------------- dispatch scatter ---
__global__ __launch_bounds__(256) void moe_scatter(const float* __restrict__ x,
                                                   const int* __restrict__ flatq,
                                                   __bf16* __restrict__ disp) {
    const int a = blockIdx.x;
    const int f = flatq[a];
    if (f < 0) return;
    const int n = a >> 1;
    const int t = threadIdx.x;
    const float4* src = (const float4*)(x + (size_t)n * MD);
    float4 u = src[t * 2], v = src[t * 2 + 1];
    v8bf o;
    o[0] = (__bf16)u.x; o[1] = (__bf16)u.y; o[2] = (__bf16)u.z; o[3] = (__bf16)u.w;
    o[4] = (__bf16)v.x; o[5] = (__bf16)v.y; o[6] = (__bf16)v.z; o[7] = (__bf16)v.w;
    *(v8bf*)(disp + (size_t)f * MD + t * 8) = o;
}

// ------------------------------------------------------------ WMMA GEMM -----
// Per expert: out[M,Nn] = act(A[M,K](bf16) @ W[K,Nn](fp32->bf16) + bias).
// 256 thr = 8 waves, block tile 128x128, wave tile 32x64 (2x4 of 16x16x32).
// Double-buffered LDS; tiles filled by async global->LDS DMA when available.
template <bool RELU, bool OUT_BF16>
__global__ __launch_bounds__(256) void moe_gemm(const __bf16* __restrict__ A,
                                                const float* __restrict__ W,
                                                const float* __restrict__ bias,
                                                void* __restrict__ outv,
                                                int M, int K, int Nn) {
    __shared__ __bf16 ldsA[2][128 * 32];   // [m][k] bf16, 8 KB per buffer
    __shared__ float  ldsB[2][32 * 128];   // [k][n] fp32, 16 KB per buffer
    const int e  = blockIdx.z;
    const int tM = blockIdx.y, tN = blockIdx.x;
    const int tid  = threadIdx.x;
    const int lane = tid & 31, wave = tid >> 5;
    const int wm = wave >> 1, wn = wave & 1;       // 4x2 wave grid
    const int hgrp = lane >> 4, mrow = lane & 15;

    const __bf16* Ae = A + (size_t)e * M * K + (size_t)(tM * 128) * K;
    const float*  We = W + (size_t)e * K * Nn + (size_t)tN * 128;
    const float*  be = bias + (size_t)e * Nn;

    v8f acc[2][4] = {};

    const int ar = tid >> 1;           // A fill: row 0..127, 32B per thread
    const int ah = (tid & 1) * 16;
    const int bk = tid >> 3;           // B fill: k row 0..31, 64B per thread
    const int bn = (tid & 7) * 16;

    auto load_tiles = [&](int k0, int buf) {
        const __bf16* ga = Ae + (size_t)ar * K + k0 + ah;
        __bf16*       la = &ldsA[buf][ar * 32 + ah];
        const float*  gb = We + (size_t)(k0 + bk) * Nn + bn;
        float*        lb = &ldsB[buf][bk * 128 + bn];
#if MOE_ASYNC
        __builtin_amdgcn_global_load_async_to_lds_b128(gcast(ga), lcast(la), 0, 0);
        __builtin_amdgcn_global_load_async_to_lds_b128(gcast(ga), lcast(la), 16, 0);
        __builtin_amdgcn_global_load_async_to_lds_b128(gcast(gb), lcast(lb), 0, 0);
        __builtin_amdgcn_global_load_async_to_lds_b128(gcast(gb), lcast(lb), 16, 0);
        __builtin_amdgcn_global_load_async_to_lds_b128(gcast(gb), lcast(lb), 32, 0);
        __builtin_amdgcn_global_load_async_to_lds_b128(gcast(gb), lcast(lb), 48, 0);
#else
        const uint4* ga4 = (const uint4*)ga;
        ((uint4*)la)[0] = ga4[0];
        ((uint4*)la)[1] = ga4[1];
        const uint4* gb4 = (const uint4*)gb;
        ((uint4*)lb)[0] = gb4[0];
        ((uint4*)lb)[1] = gb4[1];
        ((uint4*)lb)[2] = gb4[2];
        ((uint4*)lb)[3] = gb4[3];
#endif
    };

    int buf = 0;
    load_tiles(0, 0);
    for (int k0 = 0; k0 < K; k0 += 32) {
#if MOE_ASYNC
        __builtin_amdgcn_s_wait_asynccnt(0);   // own DMAs done; barrier => all
#endif
        __syncthreads();
        if (k0 + 32 < K) load_tiles(k0 + 32, buf ^ 1);   // overlap with compute

        // ---- fragments ----
        // A (16x32 bf16): lane<16 -> M=lane, K {0-7,16-23}; lane>=16 -> K {8-15,24-31}
        v16bf afr[2];
#pragma unroll
        for (int mt = 0; mt < 2; ++mt) {
            int row = wm * 32 + mt * 16 + mrow;
            int base = row * 32 + hgrp * 8;
            v8bf lo = *(const v8bf*)&ldsA[buf][base];
            v8bf hi = *(const v8bf*)&ldsA[buf][base + 16];
#pragma unroll
            for (int j = 0; j < 8; ++j) { afr[mt][j] = lo[j]; afr[mt][j + 8] = hi[j]; }
        }
        // B (32x16 bf16): lane<16 -> N=lane, K {0-7,16-23}; lane>=16 -> K {8-15,24-31}; cvt fp32->bf16 here
        v16bf bfr[4];
#pragma unroll
        for (int nt = 0; nt < 4; ++nt) {
            const float* bp = &ldsB[buf][hgrp * 8 * 128 + wn * 64 + nt * 16 + mrow];
            v16bf b;
#pragma unroll
            for (int j = 0; j < 8; ++j) { b[j] = (__bf16)bp[j * 128]; b[j + 8] = (__bf16)bp[(16 + j) * 128]; }
            bfr[nt] = b;
        }
#pragma unroll
        for (int mt = 0; mt < 2; ++mt)
#pragma unroll
            for (int nt = 0; nt < 4; ++nt)
                acc[mt][nt] = __builtin_amdgcn_wmma_f32_16x16x32_bf16(
                    false, afr[mt], false, bfr[nt], (short)0, acc[mt][nt], false, false);
        buf ^= 1;
    }

    // ---- epilogue: bias (+ReLU), C/D layout: lane<16 M=g, lane>=16 M=g+8 ----
#pragma unroll
    for (int mt = 0; mt < 2; ++mt)
#pragma unroll
        for (int nt = 0; nt < 4; ++nt) {
            int col = tN * 128 + wn * 64 + nt * 16 + mrow;
            float bv = be[col];
#pragma unroll
            for (int g = 0; g < 8; ++g) {
                int row = tM * 128 + wm * 32 + mt * 16 + hgrp * 8 + g;
                float v = acc[mt][nt][g] + bv;
                if (RELU) v = fmaxf(v, 0.f);
                size_t o = ((size_t)e * M + row) * Nn + col;
                if (OUT_BF16) ((__bf16*)outv)[o] = (__bf16)v;
                else          ((float*)outv)[o]  = v;
            }
        }
}

// --------------------------------------------------------- combine/decode ---
__global__ __launch_bounds__(256) void moe_decode(const float* __restrict__ y,
                                                  const float* __restrict__ gates,
                                                  const int* __restrict__ flatq,
                                                  float* __restrict__ out) {
    const int n = blockIdx.x, t = threadIdx.x;
    int f0 = flatq[n * 2], f1 = flatq[n * 2 + 1];
    float g0 = (f0 >= 0) ? gates[n * 2]     : 0.f;
    float g1 = (f1 >= 0) ? gates[n * 2 + 1] : 0.f;
    const float4* y0 = (const float4*)(y + (size_t)((f0 < 0) ? 0 : f0) * MD);
    const float4* y1 = (const float4*)(y + (size_t)((f1 < 0) ? 0 : f1) * MD);
    float4* o = (float4*)(out + (size_t)n * MD);
#pragma unroll
    for (int i = t; i < MD / 4; i += 256) {
        float4 a = y0[i], b = y1[i], r;
        r.x = g0 * a.x + g1 * b.x; r.y = g0 * a.y + g1 * b.y;
        r.z = g0 * a.z + g1 * b.z; r.w = g0 * a.w + g1 * b.w;
        o[i] = r;
    }
}

// ---------------------------------------------------------------- launch ----
extern "C" void kernel_launch(void* const* d_in, const int* in_sizes, int n_in,
                              void* d_out, int out_size, void* d_ws, size_t ws_size,
                              hipStream_t stream) {
    const float* x     = (const float*)d_in[0];
    const float* wg    = (const float*)d_in[1];
    const float* fc1_w = (const float*)d_in[2];
    const float* fc1_b = (const float*)d_in[3];
    const float* fc2_w = (const float*)d_in[4];
    const float* fc2_b = (const float*)d_in[5];
    float* out = (float*)d_out;

    char* ws = (char*)d_ws;
    size_t off = 0;
    auto carve = [&](size_t bytes) -> void* {
        off = (off + 255) & ~(size_t)255;
        void* p = ws + off;
        off += bytes;
        return p;
    };
    const size_t DISP_BYTES = (size_t)NE * CAP * MD * sizeof(__bf16);
    __bf16* disp   = (__bf16*)carve(DISP_BYTES);                              // 64 MB
    __bf16* h      = (__bf16*)carve((size_t)NE * CAP * HID * sizeof(__bf16)); // 128 MB
    float*  y      = (float*) carve((size_t)NE * CAP * MD * sizeof(float));   // 128 MB
    float*  scores = (float*) carve((size_t)N_TOK * NE * sizeof(float));
    int*    tki    = (int*)   carve((size_t)N_TOK * 2 * sizeof(int));
    float*  gates  = (float*) carve((size_t)N_TOK * 2 * sizeof(float));
    int*    flatq  = (int*)   carve((size_t)N_TOK * 2 * sizeof(int));
    float*  me     = (float*) carve(NE * sizeof(float));
    float*  ce     = (float*) carve(NE * sizeof(float));

    (void)hipMemsetAsync(disp, 0, DISP_BYTES, stream);

    moe_gate<<<N_TOK / 8, 256, 0, stream>>>(x, wg, scores, tki, gates);
    moe_stats<<<NE, 256, 0, stream>>>(scores, tki, me, ce);
    moe_laux<<<1, 32, 0, stream>>>(me, ce, out + (size_t)N_TOK * MD);
    moe_pos<<<NE, 256, 0, stream>>>(tki, flatq);
    moe_scatter<<<N_TOK * 2, 256, 0, stream>>>(x, flatq, disp);
    moe_gemm<true,  true ><<<dim3(HID / 128, CAP / 128, NE), 256, 0, stream>>>(
        disp, fc1_w, fc1_b, (void*)h, CAP, MD, HID);
    moe_gemm<false, false><<<dim3(MD / 128, CAP / 128, NE), 256, 0, stream>>>(
        h, fc2_w, fc2_b, (void*)y, CAP, HID, MD);
    moe_decode<<<N_TOK, 256, 0, stream>>>(y, gates, flatq, out);
}
